// EfficientSparseCODMoE_42984032698785
// MI455X (gfx1250) — matrix operand order repaired and checked by reading, not compiled, not run on hardware
//
#include <hip/hip_runtime.h>
#include <hip/hip_bf16.h>

#define B_ 16
#define C_ 128
#define H_ 96
#define W_ 96
#define E_ 6
#define HW_ (H_*W_)
#define CIP 136                       // channel pitch: 272B rows -> 16B-aligned fragments
#define XS_ELEMS (3*98*CIP)           // staged 3 rows x (96+2) cols x CIP ch (ushort)
#define HS_ELEMS (W_*CIP)             // h buffer: 96 cols x CIP ch          (ushort)

typedef __bf16        bf16x16 __attribute__((ext_vector_type(16)));
typedef float         floatx8 __attribute__((ext_vector_type(8)));
typedef unsigned int  uintx8  __attribute__((ext_vector_type(8)));
typedef unsigned int  uintx4  __attribute__((ext_vector_type(4)));

__device__ __forceinline__ unsigned int f2bf(float f) {
  unsigned int u = __float_as_uint(f);
  return (u + 0x7FFFu + ((u >> 16) & 1u)) >> 16;   // round-to-nearest-even bf16
}
__device__ __forceinline__ unsigned int packbf(float lo, float hi) {
  return f2bf(lo) | (f2bf(hi) << 16);
}
// gelu(v) = v * sigmoid(2*sqrt(2/pi)*(v + 0.044715 v^3))  (== tanh formulation)
__device__ __forceinline__ float gelu_tanh(float v) {
  const float k2 = 1.5957691216057308f;            // 2*sqrt(2/pi)
  float u = k2 * (v + 0.044715f * v * v * v);
  return v / (1.0f + __expf(-u));
}
// 16-bit A 16x32 lane layout (ISA 7.12.2): lanes0-15 K{0-7,16-23}, lanes16-31 K{8-15,24-31}
__device__ __forceinline__ int kbaseA(int j, int half) {
  return ((j & 4) << 2) + (half << 3) + ((j & 3) << 1);
}
// B 32x16 uses A's k order: lanes0-15 K{0-7,16-23}, lanes16-31 K{8-15,24-31} (two 16B reads)

// CDNA5 async global->LDS copy (ASYNCcnt); vdst = LDS byte offset, vaddr = 64-bit global
__device__ __forceinline__ void async_ld_b128(unsigned int lds_off, const void* gaddr) {
  asm volatile("global_load_async_to_lds_b128 %0, %1, off"
               :: "v"(lds_off), "v"(gaddr) : "memory");
}
__device__ __forceinline__ void wait_asynccnt0() {
  asm volatile("s_wait_asynccnt 0x0" ::: "memory");
}

// ------------------------------------------------- weight pre-pack (bf16, fragment order)
// W1 packed: uint idx = ((((e*36 + ks)*8 + wave)*32) + lane)*8 + j
__global__ void pack_w1_kernel(const float* __restrict__ W1, unsigned int* __restrict__ P) {
  int idx = blockIdx.x * 256 + threadIdx.x;
  if (idx >= E_ * 36 * 8 * 32 * 8) return;
  int j    = idx & 7;
  int lane = (idx >> 3) & 31;
  int wave = (idx >> 8) & 7;
  int kse  = idx >> 11;
  int e    = kse / 36;
  int ks   = kse - e * 36;
  int t = ks >> 2, cib = ks & 3;
  int half = lane >> 4, nlan = lane & 15;
  int co  = wave * 16 + nlan;
  int ci0 = cib * 32 + kbaseA(j, half);
  const float* w = W1 + (size_t)(e * C_ + co) * C_ * 9 + t;
  P[idx] = packbf(w[(size_t)ci0 * 9], w[(size_t)(ci0 + 1) * 9]);
}
// W2 packed: uint idx = ((((e*4 + cib)*8 + wave)*32) + lane)*8 + j
__global__ void pack_w2_kernel(const float* __restrict__ W2, unsigned int* __restrict__ P) {
  int idx = blockIdx.x * 256 + threadIdx.x;
  if (idx >= E_ * 4 * 8 * 32 * 8) return;
  int j    = idx & 7;
  int lane = (idx >> 3) & 31;
  int wave = (idx >> 8) & 7;
  int cib  = (idx >> 11) & 3;
  int e    = idx >> 13;
  int half = lane >> 4, nlan = lane & 15;
  int co  = wave * 16 + nlan;
  int ci0 = cib * 32 + kbaseA(j, half);
  const float* w = W2 + (size_t)(e * C_ + co) * C_;
  P[idx] = packbf(w[ci0], w[ci0 + 1]);
}

// ------------------------------------------- x transpose+convert: xt[b][row][col][ci] bf16
__global__ void xcvt_kernel(const float* __restrict__ x, unsigned short* __restrict__ xt) {
  __shared__ unsigned short tile[W_ * 130];      // [col][ci], pitch 130 (bank-conflict-free)
  const int row = blockIdx.x;
  const int b   = blockIdx.y;
  for (int idx = threadIdx.x; idx < C_ * W_; idx += 256) {   // read coalesced over col
    int ci = idx / W_, col = idx - ci * W_;
    float v = x[(((size_t)b * C_ + ci) * H_ + row) * W_ + col];
    tile[col * 130 + ci] = (unsigned short)f2bf(v);
  }
  __syncthreads();
  unsigned int* dst = (unsigned int*)(xt + (size_t)(b * H_ + row) * W_ * C_);
  for (int i = threadIdx.x; i < W_ * (C_ / 2); i += 256) {   // write coalesced over ci
    int col = i >> 6, cw = i & 63;
    dst[col * 64 + cw] = *(const unsigned int*)&tile[col * 130 + cw * 2];
  }
}

// ---------------------------------------------------------------- router p1
__global__ void moe_mean_kernel(const float* __restrict__ x, float* __restrict__ mean) {
  __shared__ float red[256];
  int bc = blockIdx.x;
  const float* p = x + (size_t)bc * HW_;
  float s = 0.f;
  for (int i = threadIdx.x; i < HW_; i += 256) s += p[i];
  red[threadIdx.x] = s;
  __syncthreads();
  for (int off = 128; off > 0; off >>= 1) {
    if ((int)threadIdx.x < off) red[threadIdx.x] += red[threadIdx.x + off];
    __syncthreads();
  }
  if (threadIdx.x == 0) mean[bc] = red[0] * (1.0f / HW_);
}

// ---------------------------------------------------------------- router p2
__global__ void moe_route_kernel(const float* __restrict__ mean,
                                 const float* __restrict__ Wr,
                                 const float* __restrict__ br,
                                 float* __restrict__ gate,
                                 float* __restrict__ total_out) {
  __shared__ float probs_s[B_][E_];
  __shared__ float ent_s[B_];
  int t = threadIdx.x;
  if (t < B_) {
    float logit[E_];
    for (int e = 0; e < E_; e++) {
      float a = br[e];
      for (int c = 0; c < C_; c++) a += mean[t * C_ + c] * Wr[c * E_ + e];
      logit[e] = fminf(fmaxf(a, -10.f), 10.f);
    }
    float m = logit[0];
    for (int e = 1; e < E_; e++) m = fmaxf(m, logit[e]);
    float p[E_], ps = 0.f;
    for (int e = 0; e < E_; e++) { p[e] = expf(logit[e] - m); ps += p[e]; }
    float ent = 0.f;
    for (int e = 0; e < E_; e++) {
      float pe = fminf(fmaxf(p[e] / ps, 1e-6f), 1.0f);
      p[e] = pe;
      probs_s[t][e] = pe;
      ent -= pe * logf(pe + 1e-10f);
    }
    ent_s[t] = ent;
    int i1 = 0;
    for (int e = 1; e < E_; e++) if (p[e] > p[i1]) i1 = e;      // ties -> lowest idx
    int i2 = -1;
    for (int e = 0; e < E_; e++) {
      if (e == i1) continue;
      if (i2 < 0 || p[e] > p[i2]) i2 = e;
    }
    float d = p[i1] + p[i2] + 1e-8f;
    float g1 = p[i1] / d, g2 = p[i2] / d;
    for (int e = 0; e < E_; e++)
      gate[t * E_ + e] = (e == i1) ? g1 : ((e == i2) ? g2 : 0.0f);
  }
  __syncthreads();
  if (t == 0) {
    float lb = 0.f;
    for (int e = 0; e < E_; e++) {
      float u = 0.f;
      for (int b = 0; b < B_; b++) u += probs_s[b][e];
      u *= (1.0f / B_);
      float dv = u - (1.0f / E_);
      lb += dv * dv;
    }
    float ent = 0.f;
    for (int b = 0; b < B_; b++) ent += ent_s[b];
    ent *= (1.0f / B_);
    *total_out = lb * 5e-4f - ent * 1e-3f;   // warmup_factor = 1.0
  }
}

// ------------------------------------------------------ fused expert kernel
// grid (H, B); 256 threads = 8 waves; wave w -> channels [16w,16w+16), all 96 cols
__launch_bounds__(256, 1)
__global__ void moe_expert_kernel(const float* __restrict__ x,
                                  const unsigned short* __restrict__ xt,
                                  const unsigned int* __restrict__ w1pack,
                                  const float* __restrict__ b1,
                                  const unsigned int* __restrict__ w2pack,
                                  const float* __restrict__ b2,
                                  const float* __restrict__ gate,
                                  float* __restrict__ out) {
  extern __shared__ unsigned short smem[];
  unsigned short* xs = smem;                 // [3 rows][98 cols][CIP ch] bf16
  unsigned short* hs = smem + XS_ELEMS;      // [96 cols][CIP ch] bf16

  const int r    = blockIdx.x;
  const int b    = blockIdx.y;
  const int tid  = threadIdx.x;
  const int wave = tid >> 5;
  const int lane = tid & 31;
  const int half = lane >> 4;
  const int nlan = lane & 15;
  const int mrow = wave << 4;                // this wave's channel base

  // ---- zero OOB halo rows (block-uniform branch) and halo columns
  unsigned int* xs32w = (unsigned int*)xs;
  for (int kr = 0; kr < 3; kr++) {
    int rr = r + kr - 1;
    if (rr < 0 || rr >= H_) {
      for (int i = tid; i < 98 * (CIP / 2); i += 256)
        xs32w[kr * 98 * (CIP / 2) + i] = 0;
    }
  }
  for (int i = tid; i < 3 * 2 * (C_ / 2); i += 256) {   // colp 0 and 97, all 3 rows
    int krr  = i >> 7;
    int rem2 = i & 127;
    int colp = (rem2 & 64) ? 97 : 0;
    int cw   = rem2 & 63;
    xs32w[((krr * 98 + colp) * CIP) / 2 + cw] = 0;
  }

  // ---- async-stage interior: 3 rows x 96 cols x 128ch bf16 (16B per lane transfer)
  {
    const unsigned short* xtb = xt + (size_t)b * H_ * W_ * C_;
    for (int i = tid; i < 3 * 96 * 16; i += 256) {
      int rowcol = i >> 4, lp = i & 15;
      int kr = rowcol / 96, col = rowcol - kr * 96;
      int rr = r + kr - 1;
      if (rr >= 0 && rr < H_) {
        unsigned int lds_off =
            (unsigned int)(size_t)(const void*)&xs[(kr * 98 + col + 1) * CIP + lp * 8];
        const void* gp = (const void*)(xtb + ((size_t)rr * W_ + col) * C_ + lp * 8);
        async_ld_b128(lds_off, gp);
      }
    }
  }
  wait_asynccnt0();
  __syncthreads();

  float g[E_], sgate = 0.f;
  for (int e = 0; e < E_; e++) { g[e] = gate[b * E_ + e]; sgate += g[e]; }

  floatx8 res[6];
  for (int nt = 0; nt < 6; nt++) res[nt] = (floatx8)0.0f;

  const int fraglane = wave * 32 + lane;     // fragment slot within a K-step

  for (int e = 0; e < E_; e++) {
    float ge = g[e];
    if (ge == 0.0f) continue;                // uniform per block -> EXEC stays full

    // ---------------- conv3x3 implicit GEMM: M=128, N=96, K=1152
    floatx8 acc[6];
    for (int nt = 0; nt < 6; nt++) acc[nt] = (floatx8)0.0f;
    // A fragments pre-packed: uintx8 index = (e*36 + ks)*256 + wave*32 + lane
    const uintx8* afp = ((const uintx8*)w1pack) + (size_t)e * 36 * 256 + fraglane;

    for (int t = 0; t < 9; t++) {
      int kh = t / 3, kw = t - kh * 3;
#pragma unroll
      for (int cib = 0; cib < 4; cib++) {
        uintx8 au = afp[(t * 4 + cib) * 256];
        bf16x16 afrag = __builtin_bit_cast(bf16x16, au);
#pragma unroll
        for (int nt = 0; nt < 6; nt++) {
          int colp = nt * 16 + nlan + kw;               // staged col (+1 halo offset)
          const uintx4* bp = (const uintx4*)&xs[(kh * 98 + colp) * CIP + cib * 32 + (half << 3)];
          uintx4 blo = bp[0], bhi = bp[2];
          uintx8 bu = __builtin_shufflevector(blo, bhi, 0, 1, 2, 3, 4, 5, 6, 7);
          bf16x16 bfrag = __builtin_bit_cast(bf16x16, bu);
          acc[nt] = __builtin_amdgcn_wmma_f32_16x16x32_bf16(
              false, afrag, false, bfrag, (short)0, acc[nt], false, false);
        }
      }
    }

    // bias + gelu -> hs (bf16)
    floatx8 bias1;
#pragma unroll
    for (int j = 0; j < 8; j++) bias1[j] = b1[e * C_ + mrow + (half << 3) + j];
#pragma unroll
    for (int nt = 0; nt < 6; nt++) {
      int col = nt * 16 + nlan;
#pragma unroll
      for (int j = 0; j < 8; j++) {
        float hv = gelu_tanh(acc[nt][j] + bias1[j]);
        hs[col * CIP + mrow + (half << 3) + j] = (unsigned short)f2bf(hv);
      }
    }
    __syncthreads();

    // ---------------- conv1x1 GEMM: M=128, N=96, K=128
    floatx8 acc2[6];
    for (int nt = 0; nt < 6; nt++) acc2[nt] = (floatx8)0.0f;
    const uintx8* afp2 = ((const uintx8*)w2pack) + (size_t)e * 4 * 256 + fraglane;
#pragma unroll
    for (int cib = 0; cib < 4; cib++) {
      uintx8 au = afp2[cib * 256];
      bf16x16 afrag = __builtin_bit_cast(bf16x16, au);
#pragma unroll
      for (int nt = 0; nt < 6; nt++) {
        const uintx4* bp = (const uintx4*)&hs[(nt * 16 + nlan) * CIP + cib * 32 + (half << 3)];
        uintx4 blo = bp[0], bhi = bp[2];
        uintx8 bu = __builtin_shufflevector(blo, bhi, 0, 1, 2, 3, 4, 5, 6, 7);
        bf16x16 bfrag = __builtin_bit_cast(bf16x16, bu);
        acc2[nt] = __builtin_amdgcn_wmma_f32_16x16x32_bf16(
            false, afrag, false, bfrag, (short)0, acc2[nt], false, false);
      }
    }
    floatx8 bias2;
#pragma unroll
    for (int j = 0; j < 8; j++) bias2[j] = b2[e * C_ + mrow + (half << 3) + j];
#pragma unroll
    for (int nt = 0; nt < 6; nt++) res[nt] += ge * (acc2[nt] + bias2);
    __syncthreads();                          // hs reused by next expert
  }

  // ---- epilogue: out = (sum of gates)*x + sum_e gate_e*h_e   (f32 residual)
#pragma unroll
  for (int nt = 0; nt < 6; nt++) {
    int col = nt * 16 + nlan;
#pragma unroll
    for (int j = 0; j < 8; j++) {
      int ch = mrow + (half << 3) + j;
      size_t gi = (((size_t)b * C_ + ch) * H_ + r) * W_ + col;
      out[gi] = sgate * x[gi] + res[nt][j];
    }
  }
}

// ----------------------------------------------------------------- launcher
extern "C" void kernel_launch(void* const* d_in, const int* in_sizes, int n_in,
                              void* d_out, int out_size, void* d_ws, size_t ws_size,
                              hipStream_t stream) {
  const float* x  = (const float*)d_in[0];
  const float* Wr = (const float*)d_in[1];
  const float* br = (const float*)d_in[2];
  const float* W1 = (const float*)d_in[3];
  const float* b1 = (const float*)d_in[4];
  const float* W2 = (const float*)d_in[5];
  const float* b2 = (const float*)d_in[6];
  float* out = (float*)d_out;

  // workspace layout
  unsigned short* xt = (unsigned short*)d_ws;                    // B*H*W*C bf16 (37.75 MB)
  float* mean = (float*)(xt + (size_t)B_ * H_ * W_ * C_);        // B*C floats
  float* gate = mean + B_ * C_;                                  // B*E floats (pad to 128)
  unsigned int* w1pack = (unsigned int*)(gate + 128);            // 442368 uints
  unsigned int* w2pack = w1pack + (size_t)E_ * 36 * 256 * 8;     // 49152 uints

  const int N1 = E_ * 36 * 8 * 32 * 8;
  const int N2 = E_ * 4 * 8 * 32 * 8;
  pack_w1_kernel<<<dim3((N1 + 255) / 256), dim3(256), 0, stream>>>(W1, w1pack);
  pack_w2_kernel<<<dim3((N2 + 255) / 256), dim3(256), 0, stream>>>(W2, w2pack);
  xcvt_kernel<<<dim3(H_, B_), dim3(256), 0, stream>>>(x, xt);

  moe_mean_kernel<<<dim3(B_ * C_), dim3(256), 0, stream>>>(x, mean);
  moe_route_kernel<<<dim3(1), dim3(32), 0, stream>>>(
      mean, Wr, br, gate, out + (size_t)B_ * C_ * H_ * W_);

  size_t shmem = (size_t)(XS_ELEMS + HS_ELEMS) * sizeof(unsigned short);  // ~106 KB
  moe_expert_kernel<<<dim3(H_, B_), dim3(256), shmem, stream>>>(
      x, xt, w1pack, b1, w2pack, b2, gate, out);
}
